// SimpleTransformerBlock_56633438765200
// MI455X (gfx1250) — compile-verified, matrix operand order checked by reading, not run on hardware
//
#include <hip/hip_runtime.h>
#include <stdint.h>

typedef __bf16 bf16;
typedef __attribute__((ext_vector_type(16))) __bf16 v16bf;
typedef __attribute__((ext_vector_type(8)))  __bf16 v8bf;
typedef __attribute__((ext_vector_type(8)))  float  v8f;

#define SEQL   8192
#define EMBED  768
#define HDIM   64
#define TOKENS (2*SEQL)
#define FFDIM  3072
#define LDA    40   // padded LDS row stride for 32-wide K slabs (bf16 elems)
#define LDK    72   // padded LDS row stride for 64-wide K rows (bf16 elems)

// Concatenate two 8-wide bf16 vectors into a 16-wide WMMA operand.
__device__ __forceinline__ v16bf cat16(v8bf a, v8bf b){
  return __builtin_shufflevector(a,b,0,1,2,3,4,5,6,7,8,9,10,11,12,13,14,15);
}

// Load one 16x32 A/B fragment row for this lane (row ptr selected by lane%16).
// CDNA5 16-bit operand layout: lanes 0-15 hold K = {0..7, 16..23},
// lanes 16-31 hold K = {8..15, 24..31}  ->  two 16B chunks per lane.
__device__ __forceinline__ v16bf ldfrag(const bf16* row, int koff, int half){
  v8bf a = *(const v8bf*)(row + koff +      half*8);
  v8bf b = *(const v8bf*)(row + koff + 16 + half*8);
  return cat16(a,b);
}

__device__ __forceinline__ v8f wmma_bf16(v16bf a, v16bf b, v8f c){
  return __builtin_amdgcn_wmma_f32_16x16x32_bf16(false, a, false, b, (short)0, c, false, false);
}

// ---------------------------------------------------------------------------
// CDNA5 async global->LDS DMA (ASYNCcnt-tracked). LDS byte offset = low 32
// bits of the generic (flat) address of a __shared__ object: the LDS aperture
// places the intra-LDS offset in addr[31:0] (ISA 10.2 aperture rules).
// ---------------------------------------------------------------------------
__device__ __forceinline__ uint32_t lds_off(const void* p){
  return (uint32_t)(uintptr_t)p;
}
__device__ __forceinline__ void async_b128(uint32_t lds, const void* gaddr){
  asm volatile("global_load_async_to_lds_b128 %0, %1, off"
               :: "v"(lds), "v"(gaddr) : "memory");
}
__device__ __forceinline__ void wait_async0(){
  asm volatile("s_wait_asynccnt 0x0" ::: "memory");
}

// ---------------------------------------------------------------------------
// Weight convert + transpose: W[K][N] fp32 -> Wt[N][K] bf16
// ---------------------------------------------------------------------------
__global__ void wconv_kernel(const float* __restrict__ W, bf16* __restrict__ Wt,
                             int K, int N){
  int idx = blockIdx.x*256 + threadIdx.x;
  if (idx < K*N){
    int k = idx / N, n = idx % N;
    Wt[(size_t)n*K + k] = (bf16)W[idx];
  }
}

// ---------------------------------------------------------------------------
// Zero fill (attention scatter target must start at 0)
// ---------------------------------------------------------------------------
__global__ void zfill_kernel(uint32_t* __restrict__ p, size_t n){
  size_t i  = (size_t)blockIdx.x*blockDim.x + threadIdx.x;
  size_t st = (size_t)gridDim.x*blockDim.x;
  for (; i < n; i += st) p[i] = 0u;
}

// ---------------------------------------------------------------------------
// LayerNorm over E=768, one token per block, bf16 output for GEMM consumption
// ---------------------------------------------------------------------------
__global__ __launch_bounds__(256) void ln_to_bf16(const float* __restrict__ X,
    const float* __restrict__ g, const float* __restrict__ bta,
    bf16* __restrict__ H){
  int tok = blockIdx.x, t = threadIdx.x;
  const float* xr = X + (size_t)tok*EMBED;
  float a0 = xr[t], a1 = xr[t+256], a2 = xr[t+512];
  __shared__ float r1[256], r2[256];
  r1[t] = a0+a1+a2;
  r2[t] = a0*a0 + a1*a1 + a2*a2;
  __syncthreads();
  for (int st=128; st>0; st>>=1){
    if (t < st){ r1[t] += r1[t+st]; r2[t] += r2[t+st]; }
    __syncthreads();
  }
  float mu  = r1[0]*(1.0f/EMBED);
  float var = r2[0]*(1.0f/EMBED) - mu*mu;
  float rs  = rsqrtf(var + 1e-5f);
  bf16* hr = H + (size_t)tok*EMBED;
  hr[t    ] = (bf16)((a0-mu)*rs*g[t    ] + bta[t    ]);
  hr[t+256] = (bf16)((a1-mu)*rs*g[t+256] + bta[t+256]);
  hr[t+512] = (bf16)((a2-mu)*rs*g[t+512] + bta[t+512]);
}

// ---------------------------------------------------------------------------
// WMMA GEMM: C[M,N] = A_bf16[M,K] x Bt_bf16[N,K]^T + bias (+gelu) (+resid)
// Block tile 128x256, 8 waves (2x4), wave tile 64x64 (16 WMMA / 8 frag-loads),
// BK = 32 (one WMMA k-step), double-buffered LDS, staging via CDNA5 async
// global->LDS DMA: prefetch is fire-and-forget while WMMAs issue; one
// s_wait_asynccnt + one barrier per k-step.
// ---------------------------------------------------------------------------
__global__ __launch_bounds__(256) void gemm_bf16(
    const bf16*  __restrict__ A,    const bf16*  __restrict__ Bt,
    const float* __restrict__ bias, const float* __restrict__ resid,
    float* __restrict__ outF, bf16* __restrict__ outB,
    int M, int N, int K, int gelu)
{
  __shared__ alignas(16) bf16 As[2*128*LDA];
  __shared__ alignas(16) bf16 Bs[2*256*LDA];
  int mbase = blockIdx.y*128, nbase = blockIdx.x*256;
  int t = threadIdx.x, w = t>>5, lane = t&31, half = lane>>4, lm = lane&15;
  int wM = (w>>2)*64, wN = (w&3)*64;
  v8f acc[4][4] = {};
  int ar = t>>1, ac = (t&1)*16;

  // per-thread staging addresses (buffer 0); buffer 1 = +byte delta
  const uint32_t aoff = lds_off(&As[ar*LDA + ac]);
  const uint32_t boff = lds_off(&Bs[t*LDA]);
  const uint32_t adel = 128*LDA*2;   // bytes between A buffers
  const uint32_t bdel = 256*LDA*2;   // bytes between B buffers
  const bf16* agp = A  + (size_t)(mbase+ar)*K + ac;
  const bf16* bgp = Bt + (size_t)(nbase+t)*K;

  // stage K-slab 0 into buffer 0 (async DMA, 6 x 16B per thread)
  async_b128(aoff,      agp);
  async_b128(aoff + 16, agp + 8);
  async_b128(boff,      bgp);
  async_b128(boff + 16, bgp + 8);
  async_b128(boff + 32, bgp + 16);
  async_b128(boff + 48, bgp + 24);
  wait_async0();
  __syncthreads();

  int buf = 0;
  for (int k0 = 0; k0 < K; k0 += 32){
    bool more = (k0 + 32) < K;
    if (more){ // fire-and-forget prefetch of next slab into the other buffer
      uint32_t a2 = aoff + (uint32_t)(buf^1)*adel;
      uint32_t b2 = boff + (uint32_t)(buf^1)*bdel;
      const bf16* ag = agp + k0 + 32;
      const bf16* bg = bgp + k0 + 32;
      async_b128(a2,      ag);
      async_b128(a2 + 16, ag + 8);
      async_b128(b2,      bg);
      async_b128(b2 + 16, bg + 8);
      async_b128(b2 + 32, bg + 16);
      async_b128(b2 + 48, bg + 24);
    }

    const bf16* Ab = &As[buf*128*LDA];
    const bf16* Bb = &Bs[buf*256*LDA];
    v16bf bfg[4];
    #pragma unroll
    for (int nt=0; nt<4; nt++)
      bfg[nt] = ldfrag(&Bb[(wN + nt*16 + lm)*LDA], 0, half);
    #pragma unroll
    for (int mt=0; mt<4; mt++){
      v16bf af = ldfrag(&Ab[(wM + mt*16 + lm)*LDA], 0, half);
      #pragma unroll
      for (int nt=0; nt<4; nt++)
        acc[mt][nt] = wmma_bf16(af, bfg[nt], acc[mt][nt]);
    }

    if (more) wait_async0();
    __syncthreads();
    buf ^= 1;
  }

  #pragma unroll
  for (int mt=0; mt<4; mt++)
    #pragma unroll
    for (int nt=0; nt<4; nt++)
      #pragma unroll
      for (int v=0; v<8; v++){
        int row = mbase + wM + mt*16 + v + 8*half;   // D layout: M = v + 8*(lane/16)
        int col = nbase + wN + nt*16 + lm;           //           N = lane%16
        float val = acc[mt][nt][v] + bias[col];
        if (gelu) val = 0.5f*val*(1.0f + erff(val*0.70710678118f));
        size_t o = (size_t)row*N + col;
        if (resid) val += resid[o];
        if (outF)  outF[o] = val;
        if (outB)  outB[o] = (bf16)val;
      }
}

// ---------------------------------------------------------------------------
// Flash-style dilated attention.
// Per block: one (batch, group, segment, head, 128-query chunk).
// Per wave: 16 queries. Sᵀ = K·Qᵀ (lane = query → in-lane softmax);
// the two stacked Sᵀ D-tiles are already the B-operand layout of Pᵀ,
// so Oᵀ += Vᵀ·Pᵀ needs no data movement between matmuls.
// K tile staged row-major via async DMA; V tile staged transposed in LDS.
// ---------------------------------------------------------------------------
__global__ __launch_bounds__(256) void dilated_attn(
    const bf16* __restrict__ Q, const bf16* __restrict__ Kb,
    const bf16* __restrict__ V, bf16* __restrict__ O)
{
  __shared__ alignas(16) bf16 Ks[32*LDK];   // K tile:  32 keys x 64 dd (row-major)
  __shared__ alignas(16) bf16 VT[64*LDA];   // Vᵀ tile: 64 dd   x 32 keys

  int tmp = blockIdx.x;
  int qc = tmp & 15; tmp >>= 4;
  int hh = tmp & 3;  tmp >>= 2;
  int segidx = tmp % 7; int b = tmp / 7;
  int i, seg;
  if      (segidx < 4){ i = 0; seg = segidx;     }
  else if (segidx < 6){ i = 1; seg = segidx - 4; }
  else                { i = 2; seg = 0;          }
  int s, r;
  if      (i == 0){ s = 2048; r = 1; }
  else if (i == 1){ s = 4096; r = 2; }
  else            { s = 8192; r = 4; }
  int off  = i % r;
  int head = i*4 + hh;

  size_t base = (size_t)b*SEQL + (size_t)seg*s + off;  // token = base + j*r

  int t = threadIdx.x, wave = t>>5, lane = t&31, half = lane>>4, lm = lane&15;
  int j = qc*128 + wave*16 + lm;                       // query (dilated index)

  const bf16* qrow = Q + (base + (size_t)j*r)*EMBED + head*HDIM;
  v16bf bq0 = ldfrag(qrow, 0,  half);                  // Qᵀ B-frags, dd 0..31
  v16bf bq1 = ldfrag(qrow, 32, half);                  //              dd 32..63

  v8f o0 = {}, o1 = {}, o2 = {}, o3 = {};
  float mrun = -3.0e38f, lrun = 0.0f;
  int kk = t>>3, c = (t&7)*8;                          // staging coords
  const uint32_t ksOff = lds_off(&Ks[kk*LDK + c]);

  for (int kb = 0; kb < 2048; kb += 32){
    __syncthreads();
    { // stage K tile (async DMA) and Vᵀ tile (transposed scalar stores)
      size_t rowoff = (base + (size_t)(kb+kk)*r)*EMBED + head*HDIM + c;
      async_b128(ksOff, Kb + rowoff);
      v8bf vv = *(const v8bf*)(V + rowoff);
      #pragma unroll
      for (int u=0; u<8; u++) VT[(c+u)*LDA + kk] = vv[u];
    }
    wait_async0();
    __syncthreads();

    // Sᵀ tiles: A = K rows (16 keys each), B = Qᵀ; chain over dd = 2 k-steps
    v8f z = {};
    v8f s0 = wmma_bf16(ldfrag(&Ks[(     lm)*LDK], 0, half), bq0, z);
    s0     = wmma_bf16(ldfrag(&Ks[(     lm)*LDK],32, half), bq1, s0);
    v8f s1 = wmma_bf16(ldfrag(&Ks[(16 + lm)*LDK], 0, half), bq0, z);
    s1     = wmma_bf16(ldfrag(&Ks[(16 + lm)*LDK],32, half), bq1, s1);

    // online softmax (per query = per lane; other key-half via shfl_xor 16)
    float mx = -3.0e38f;
    #pragma unroll
    for (int v=0; v<8; v++){
      s0[v] *= 0.125f; s1[v] *= 0.125f;                // 1/sqrt(64)
      mx = fmaxf(mx, fmaxf(s0[v], s1[v]));
    }
    mx = fmaxf(mx, __shfl_xor(mx, 16, 32));
    float mnew  = fmaxf(mrun, mx);
    float alpha = __expf(mrun - mnew);
    float psum  = 0.0f;
    v16bf pv;                                          // Pᵀ already in B layout
    #pragma unroll
    for (int v=0; v<8; v++){
      float p0 = __expf(s0[v] - mnew);
      float p1 = __expf(s1[v] - mnew);
      psum += p0 + p1;
      pv[v]   = (bf16)p0;
      pv[v+8] = (bf16)p1;
    }
    psum += __shfl_xor(psum, 16, 32);
    lrun = lrun*alpha + psum;
    mrun = mnew;
    #pragma unroll
    for (int v=0; v<8; v++){ o0[v]*=alpha; o1[v]*=alpha; o2[v]*=alpha; o3[v]*=alpha; }

    // Oᵀ += Vᵀ(dd-tile) x Pᵀ
    o0 = wmma_bf16(ldfrag(&VT[( 0 + lm)*LDA], 0, half), pv, o0);
    o1 = wmma_bf16(ldfrag(&VT[(16 + lm)*LDA], 0, half), pv, o1);
    o2 = wmma_bf16(ldfrag(&VT[(32 + lm)*LDA], 0, half), pv, o2);
    o3 = wmma_bf16(ldfrag(&VT[(48 + lm)*LDA], 0, half), pv, o3);
  }

  float inv = 1.0f / lrun;
  bf16* orow = O + (base + (size_t)j*r)*EMBED + head*HDIM;
  #pragma unroll
  for (int v=0; v<8; v++){
    orow[ 0 + v + 8*half] = (bf16)(o0[v]*inv);
    orow[16 + v + 8*half] = (bf16)(o1[v]*inv);
    orow[32 + v + 8*half] = (bf16)(o2[v]*inv);
    orow[48 + v + 8*half] = (bf16)(o3[v]*inv);
  }
}

// ---------------------------------------------------------------------------
// Launch: full transformer block pipeline
// ---------------------------------------------------------------------------
extern "C" void kernel_launch(void* const* d_in, const int* in_sizes, int n_in,
                              void* d_out, int out_size, void* d_ws, size_t ws_size,
                              hipStream_t stream) {
  const float* x   = (const float*)d_in[0];
  const float* Wq  = (const float*)d_in[1];  const float* bq = (const float*)d_in[2];
  const float* Wk  = (const float*)d_in[3];  const float* bk = (const float*)d_in[4];
  const float* Wv  = (const float*)d_in[5];  const float* bv = (const float*)d_in[6];
  const float* Wo  = (const float*)d_in[7];  const float* bo = (const float*)d_in[8];
  const float* g1  = (const float*)d_in[9];  const float* be1= (const float*)d_in[10];
  const float* g2  = (const float*)d_in[11]; const float* be2= (const float*)d_in[12];
  const float* W1  = (const float*)d_in[13]; const float* b1 = (const float*)d_in[14];
  const float* W2  = (const float*)d_in[15]; const float* b2 = (const float*)d_in[16];

  char* p = (char*)d_ws;
  auto wsAlloc = [&](size_t bytes)->void*{
    void* r = p; p += (bytes + 255) & ~(size_t)255; return r;
  };
  bf16*  hB    = (bf16*) wsAlloc((size_t)TOKENS*EMBED*2);
  bf16*  qB    = (bf16*) wsAlloc((size_t)TOKENS*EMBED*2);
  bf16*  kB    = (bf16*) wsAlloc((size_t)TOKENS*EMBED*2);
  bf16*  vB    = (bf16*) wsAlloc((size_t)TOKENS*EMBED*2);
  bf16*  attnB = (bf16*) wsAlloc((size_t)TOKENS*EMBED*2);
  float* x1    = (float*)wsAlloc((size_t)TOKENS*EMBED*4);
  bf16*  h2B   = (bf16*) wsAlloc((size_t)TOKENS*EMBED*2);
  bf16*  mB    = (bf16*) wsAlloc((size_t)TOKENS*FFDIM*2);
  bf16*  WqT   = (bf16*) wsAlloc((size_t)EMBED*EMBED*2);
  bf16*  WkT   = (bf16*) wsAlloc((size_t)EMBED*EMBED*2);
  bf16*  WvT   = (bf16*) wsAlloc((size_t)EMBED*EMBED*2);
  bf16*  WoT   = (bf16*) wsAlloc((size_t)EMBED*EMBED*2);
  bf16*  W1T   = (bf16*) wsAlloc((size_t)EMBED*FFDIM*2);
  bf16*  W2T   = (bf16*) wsAlloc((size_t)FFDIM*EMBED*2);

  int nEE = EMBED*EMBED, nEF = EMBED*FFDIM;
  wconv_kernel<<<(nEE+255)/256,256,0,stream>>>(Wq, WqT, EMBED, EMBED);
  wconv_kernel<<<(nEE+255)/256,256,0,stream>>>(Wk, WkT, EMBED, EMBED);
  wconv_kernel<<<(nEE+255)/256,256,0,stream>>>(Wv, WvT, EMBED, EMBED);
  wconv_kernel<<<(nEE+255)/256,256,0,stream>>>(Wo, WoT, EMBED, EMBED);
  wconv_kernel<<<(nEF+255)/256,256,0,stream>>>(W1, W1T, EMBED, FFDIM);
  wconv_kernel<<<(nEF+255)/256,256,0,stream>>>(W2, W2T, FFDIM, EMBED);

  // LN1 -> h (bf16)
  ln_to_bf16<<<TOKENS,256,0,stream>>>(x, g1, be1, hB);

  // QKV projections (bf16 outputs feed attention)
  dim3 gEE(EMBED/256, TOKENS/128);
  gemm_bf16<<<gEE,256,0,stream>>>(hB, WqT, bq, nullptr, nullptr, qB, TOKENS, EMBED, EMBED, 0);
  gemm_bf16<<<gEE,256,0,stream>>>(hB, WkT, bk, nullptr, nullptr, kB, TOKENS, EMBED, EMBED, 0);
  gemm_bf16<<<gEE,256,0,stream>>>(hB, WvT, bv, nullptr, nullptr, vB, TOKENS, EMBED, EMBED, 0);

  // attention scatter target must be zero (non-selected dilated slots stay 0)
  zfill_kernel<<<2048,256,0,stream>>>((uint32_t*)attnB, (size_t)TOKENS*EMBED*2/4);

  // 2 batches * 7 segments * 4 heads * 16 query-chunks = 896 blocks
  dilated_attn<<<896,256,0,stream>>>(qB, kB, vB, attnB);

  // O-projection + residual(x) -> x1 (fp32)
  gemm_bf16<<<gEE,256,0,stream>>>(attnB, WoT, bo, x, x1, nullptr, TOKENS, EMBED, EMBED, 0);

  // LN2 -> h2 (bf16)
  ln_to_bf16<<<TOKENS,256,0,stream>>>(x1, g2, be2, h2B);

  // MLP: GELU(h2 @ W1 + b1) @ W2 + b2 + x1 -> d_out
  dim3 gEF(FFDIM/256, TOKENS/128);
  gemm_bf16<<<gEF,256,0,stream>>>(h2B, W1T, b1, nullptr, nullptr, mB, TOKENS, FFDIM, EMBED, 1);
  dim3 gFE(EMBED/256, TOKENS/128);
  gemm_bf16<<<gFE,256,0,stream>>>(mB, W2T, b2, x1, (float*)d_out, nullptr, TOKENS, EMBED, FFDIM, 0);
}
